// Baseline_12927851561574
// MI455X (gfx1250) — compile-verified, matrix operand order checked by reading, not run on hardware
//
#include <hip/hip_runtime.h>

typedef __attribute__((ext_vector_type(16))) __bf16 v16bf;
typedef __attribute__((ext_vector_type(8)))  float  v8f;

union AFrag { unsigned u[8]; v16bf v; };

__device__ __forceinline__ unsigned short f2bf(float f) {
  return __builtin_bit_cast(unsigned short, (__bf16)f);
}

#if __has_builtin(__builtin_amdgcn_tanhf)
__device__ __forceinline__ float tanh_(float x) { return __builtin_amdgcn_tanhf(x); }
#else
__device__ __forceinline__ float tanh_(float x) {
  float e = __expf(-2.0f * fabsf(x));
  float t = (1.0f - e) / (1.0f + e);
  return x < 0.0f ? -t : t;
}
#endif
__device__ __forceinline__ float sigf(float x) { return 0.5f * tanh_(0.5f * x) + 0.5f; }

#define TT 72
#define NB 8192

// ---------------------------------------------------------------------------
// Prep: build bf16 B-fragment-swizzled concatenated weights [Wih | Whh | pad]
// wfrag layout: [enc][j(16)][kc(4)][lane(32)][v(8)] packed bf16 pairs (uint).
// ---------------------------------------------------------------------------
__global__ void prep_kernel(const float* oWih, const float* oWhh, const float* obih, const float* obhh,
                            const float* wWih, const float* wWhh, const float* wbih, const float* wbhh,
                            unsigned* wfrag, float* bias) {
  const int e   = blockIdx.x;
  const int tid = threadIdx.x;
  const float* Wih = e ? wWih : oWih;
  const float* Whh = e ? wWhh : oWhh;
  const float* bih = e ? wbih : obih;
  const float* bhh = e ? wbhh : obhh;
  const int C = e ? 56 : 32;
  unsigned* wf = wfrag + e * 16384;

  for (int idx = tid; idx < 16384; idx += blockDim.x) {
    int v    = idx & 7;
    int lane = (idx >> 3) & 31;
    int kc   = (idx >> 8) & 3;
    int j    = idx >> 10;
    int hf   = lane >> 4;
    int n    = lane & 15;
    int g    = 16 * j + n;                                   // gate output index 0..255
    int k0   = kc * 32 + ((v < 4) ? 0 : 16) + hf * 8 + 2 * (v & 3);
    float f0 = 0.0f, f1 = 0.0f;
    int k = k0;
    if (k < C) f0 = Wih[g * C + k]; else if (k < C + 64) f0 = Whh[g * 64 + (k - C)];
    k = k0 + 1;
    if (k < C) f1 = Wih[g * C + k]; else if (k < C + 64) f1 = Whh[g * 64 + (k - C)];
    wf[idx] = ((unsigned)f2bf(f1) << 16) | (unsigned)f2bf(f0);
  }
  for (int gi = tid; gi < 256; gi += blockDim.x) bias[e * 256 + gi] = bih[gi] + bhh[gi];
}

// ---------------------------------------------------------------------------
// Fused LSTM body (compile-time C / NKC). One wave owns 16 batch rows for the
// whole sequence; all B fragments + biases live in VGPRs across the time loop.
// x_t is software-pipelined through registers so the per-step critical path
// never waits on global memory.
// ---------------------------------------------------------------------------
template <int C, int NKC>
__device__ __forceinline__ void lstm_body(const float* __restrict__ X,
                                          const unsigned* __restrict__ wf,
                                          const float* __restrict__ biasg,
                                          float* __restrict__ feat,
                                          unsigned short* xw_, int n0, int lane, int enc) {
  const int hf  = lane >> 4;
  const int mlo = lane & 15;
  constexpr int NLD = (16 * C) / 32;   // per-lane x elements per step (exact: 16, 28)

  // loop-invariant per-lane bias scalars: bsv[j] covers gate cols 16j..16j+15
  float bsv[16];
  #pragma unroll
  for (int j = 0; j < 16; ++j) bsv[j] = biasg[enc * 256 + 16 * j + mlo];

  // loop-invariant B fragments, kept in VGPRs for the entire sequence
  AFrag bfr[16][NKC];
  #pragma unroll
  for (int j = 0; j < 16; ++j) {
    #pragma unroll
    for (int kc = 0; kc < NKC; ++kc) {
      const unsigned* wp = wf + (size_t)((j * 4 + kc) * 32 + lane) * 8;
      #pragma unroll
      for (int v = 0; v < 8; ++v) bfr[j][kc].u[v] = wp[v];
    }
  }

  v8f cst[4];
  #pragma unroll
  for (int hj = 0; hj < 4; ++hj)
    #pragma unroll
    for (int r = 0; r < 8; ++r) cst[hj][r] = 0.0f;

  // prime the x pipeline with t = 0
  float xreg[NLD];
  #pragma unroll
  for (int i = 0; i < NLD; ++i) {
    int idx = lane + 32 * i;
    int r = idx / C, c = idx - r * C;
    xreg[i] = X[((size_t)(n0 + r) * C + c) * TT];
  }

  for (int t = 0; t < TT; ++t) {
    // commit x_t (already in registers) to the LDS A-staging buffer
    #pragma unroll
    for (int i = 0; i < NLD; ++i) {
      int idx = lane + 32 * i;
      int r = idx / C, c = idx - r * C;
      xw_[r * 128 + c] = f2bf(xreg[i]);
    }

    // issue loads for x_{t+1}; consumed only at the end of this step
    float xnext[NLD];
    if (t + 1 < TT) {
      #pragma unroll
      for (int i = 0; i < NLD; ++i) {
        int idx = lane + 32 * i;
        int r = idx / C, c = idx - r * C;
        xnext[i] = X[((size_t)(n0 + r) * C + c) * TT + (t + 1)];
      }
    } else {
      #pragma unroll
      for (int i = 0; i < NLD; ++i) xnext[i] = 0.0f;
    }

    // A fragments (16x32 bf16 layout) from LDS
    AFrag a[NKC];
    #pragma unroll
    for (int kc = 0; kc < NKC; ++kc) {
      const unsigned* p = (const unsigned*)(xw_ + mlo * 128);
      #pragma unroll
      for (int v = 0; v < 8; ++v) {
        int k = kc * 32 + ((v < 4) ? 0 : 16) + hf * 8 + 2 * (v & 3);
        a[kc].u[v] = p[k >> 1];
      }
    }

    #pragma unroll
    for (int hj = 0; hj < 4; ++hj) {           // hidden tile (16 units each)
      v8f gg[4];                               // i, f, g, o accumulators
      #pragma unroll
      for (int gs = 0; gs < 4; ++gs) {
        const int j = hj + 4 * gs;             // column tile in gates[.,256]
        v8f acc;
        #pragma unroll
        for (int r = 0; r < 8; ++r) acc[r] = bsv[j];
        #pragma unroll
        for (int kc = 0; kc < NKC; ++kc) {
          acc = __builtin_amdgcn_wmma_f32_16x16x32_bf16(
              false, a[kc].v, false, bfr[j][kc].v, (short)0, acc, false, false);
        }
        gg[gs] = acc;
      }
      // elementwise LSTM cell on C/D layout: VGPR r, lane -> (M = r or r+8, n = lane&15)
      #pragma unroll
      for (int r = 0; r < 8; ++r) {
        float iv = sigf(gg[0][r]);
        float fv = sigf(gg[1][r]);
        float gv = tanh_(gg[2][r]);
        float ov = sigf(gg[3][r]);
        float cn = fv * cst[hj][r] + iv * gv;
        cst[hj][r] = cn;
        float hv = ov * tanh_(cn);
        int m = hf ? (r + 8) : r;
        xw_[m * 128 + C + hj * 16 + mlo] = f2bf(hv);     // h feeds next step
        if (t == TT - 1)
          feat[(size_t)(n0 + m) * 128 + enc * 64 + hj * 16 + mlo] = hv;
      }
    }

    // rotate the x pipeline (forces the loadcnt wait here, after all WMMA issue)
    #pragma unroll
    for (int i = 0; i < NLD; ++i) xreg[i] = xnext[i];
  }
}

__global__ void __launch_bounds__(128)
lstm_kernel(const float* __restrict__ Xo, const float* __restrict__ Xw,
            const unsigned* __restrict__ wfrag, const float* __restrict__ biasg,
            float* __restrict__ feat) {
  __shared__ unsigned short xh[4 * 16 * 128];   // per-wave [16 rows][128 K] bf16 A staging

  const int tid  = threadIdx.x;
  const int w    = tid >> 5;
  const int lane = tid & 31;
  const int enc  = blockIdx.y;
  const int n0   = (blockIdx.x * 4 + w) * 16;

  // zero own staging region (pads + initial h = 0); wave-private, no barrier needed
  unsigned* xz = (unsigned*)(xh + w * 16 * 128);
  for (int i = lane; i < 1024; i += 32) xz[i] = 0u;

  unsigned short* xw_ = xh + w * 16 * 128;

  if (enc == 0)
    lstm_body<32, 3>(Xo, wfrag,         biasg, feat, xw_, n0, lane, 0);
  else
    lstm_body<56, 4>(Xw, wfrag + 16384, biasg, feat, xw_, n0, lane, 1);
}

// ---------------------------------------------------------------------------
// Heads: feat[N,128] -> relu(W1)->relu(W2)->W3, both heads, out [N,2,48].
// ---------------------------------------------------------------------------
__global__ void __launch_bounds__(256)
heads_kernel(const float* __restrict__ feat,
             const float* fW1, const float* fb1, const float* fW2, const float* fb2,
             const float* fW3, const float* fb3,
             const float* oW1, const float* ob1, const float* oW2, const float* ob2,
             const float* oW3, const float* ob3,
             float* __restrict__ out) {
  __shared__ float smem[8192 + 6144];  // ftile [64][128] (reused as t2 [64][64]) + t1 [64][96]
  float* ftile = smem;
  float* t1    = smem + 8192;
  const int tid = threadIdx.x;
  const int r0  = blockIdx.x * 64;

  for (int hsel = 0; hsel < 2; ++hsel) {
    const float* W1 = hsel ? oW1 : fW1;  const float* b1 = hsel ? ob1 : fb1;
    const float* W2 = hsel ? oW2 : fW2;  const float* b2 = hsel ? ob2 : fb2;
    const float* W3 = hsel ? oW3 : fW3;  const float* b3 = hsel ? ob3 : fb3;

    __syncthreads();  // protect t2 region from previous iteration
    for (int i = tid; i < 64 * 128; i += 256) ftile[i] = feat[(size_t)r0 * 128 + i];
    __syncthreads();

    for (int o = tid; o < 64 * 96; o += 256) {
      int r = o / 96, k = o - r * 96;
      const float* wr = W1 + k * 128;
      const float* fr = ftile + r * 128;
      float s = b1[k];
      #pragma unroll 8
      for (int c = 0; c < 128; ++c) s += fr[c] * wr[c];
      t1[o] = s > 0.0f ? s : 0.0f;
    }
    __syncthreads();

    float* t2 = smem;  // overwrite ftile (done with it)
    for (int o = tid; o < 64 * 64; o += 256) {
      int r = o >> 6, k = o & 63;
      const float* wr = W2 + k * 96;
      const float* ar = t1 + r * 96;
      float s = b2[k];
      #pragma unroll 8
      for (int c = 0; c < 96; ++c) s += ar[c] * wr[c];
      t2[o] = s > 0.0f ? s : 0.0f;
    }
    __syncthreads();

    for (int o = tid; o < 64 * 48; o += 256) {
      int r = o / 48, k = o - r * 48;
      const float* wr = W3 + k * 64;
      const float* ar = t2 + r * 64;
      float s = b3[k];
      #pragma unroll 8
      for (int c = 0; c < 64; ++c) s += ar[c] * wr[c];
      out[(size_t)(r0 + r) * 96 + hsel * 48 + k] = s;
    }
  }
}

extern "C" void kernel_launch(void* const* d_in, const int* in_sizes, int n_in,
                              void* d_out, int out_size, void* d_ws, size_t ws_size,
                              hipStream_t stream) {
  const float* X_obs = (const float*)d_in[0];
  const float* X_wrf = (const float*)d_in[1];

  // workspace layout
  unsigned* wfrag = (unsigned*)d_ws;                         // 131072 B
  float* bias     = (float*)((char*)d_ws + 131072);          //   2048 B
  float* feat     = (float*)((char*)d_ws + 133120);          // 4 MB [N][128]

  prep_kernel<<<dim3(2), 256, 0, stream>>>(
      (const float*)d_in[2], (const float*)d_in[3], (const float*)d_in[4], (const float*)d_in[5],
      (const float*)d_in[6], (const float*)d_in[7], (const float*)d_in[8], (const float*)d_in[9],
      wfrag, bias);

  lstm_kernel<<<dim3(NB / 64, 2), 128, 0, stream>>>(X_obs, X_wrf, wfrag, bias, feat);

  heads_kernel<<<dim3(NB / 64), 256, 0, stream>>>(
      feat,
      (const float*)d_in[10], (const float*)d_in[11], (const float*)d_in[12],
      (const float*)d_in[13], (const float*)d_in[14], (const float*)d_in[15],
      (const float*)d_in[16], (const float*)d_in[17], (const float*)d_in[18],
      (const float*)d_in[19], (const float*)d_in[20], (const float*)d_in[21],
      (float*)d_out);
}
